// Dual_Block_51281909514646
// MI455X (gfx1250) — matrix operand order repaired and checked by reading, not compiled, not run on hardware
//
#include <hip/hip_runtime.h>
#include <stdint.h>

// ---------------------------------------------------------------------------
// Problem constants (from reference): N=8, C=3, H=W=256; RDN G0=64,G=32,D=2,CL=3
// ---------------------------------------------------------------------------
#define SPA   65536          // 256*256 spatial
#define NB    8
#define ETA0f 1.0f
#define ETA1f 0.7f
#define ETA2f 0.3f
#define ETA3f 1.0f
#define ETA4f 1.0f

typedef __attribute__((ext_vector_type(16))) __bf16 bf16x16;
typedef __attribute__((ext_vector_type(8)))  float  floatx8;

union U16x16 { unsigned short u[16]; bf16x16 v; };
struct u32x4 { unsigned int x, y, z, w; };   // POD 16B for wide weight loads

__device__ __forceinline__ unsigned short f32_to_bf16(float f) {
  unsigned int u = __float_as_uint(f);
  u += 0x7FFFu + ((u >> 16) & 1u);           // round-to-nearest-even
  return (unsigned short)(u >> 16);
}
__device__ __forceinline__ float bf16_to_f32(unsigned short h) {
  return __uint_as_float(((unsigned int)h) << 16);
}
__device__ __forceinline__ int imin(int a, int b) { return a < b ? a : b; }
__device__ __forceinline__ int imax(int a, int b) { return a > b ? a : b; }

// Low 32 bits of a generic pointer to LDS == byte offset within LDS.
__device__ __forceinline__ unsigned int lds_off32(const void* p) {
  return (unsigned int)(uintptr_t)p;
}

// ---------------------------------------------------------------------------
// Weight repack: OIHW f32 -> bf16 WMMA A-fragments in r-major K order.
// Logical k = r*CinPad + ci  (r = kh*3+kw), so each 32-K chunk is one (kh,kw)
// with 32 contiguous input channels.  A 16x32 tile lane layout:
//   lanes 0-15 (M=lane):    VGPR0-3 K=0..7,  VGPR4-7 K=16..23
//   lanes 16-31 (M=lane-16):VGPR0-3 K=8..15, VGPR4-7 K=24..31   (2 bf16/dword)
// dst dword index = ((coTile*Ktiles + kt)*32 + lane)*8 + j
// ---------------------------------------------------------------------------
__global__ void repack_w(const float* __restrict__ W, int Cout, int CinReal,
                         int CinPad, int ksq, unsigned int* __restrict__ dst,
                         int total) {
  int idx = blockIdx.x * 256 + threadIdx.x;
  if (idx >= total) return;
  int Ktiles = (CinPad * ksq) >> 5;
  int j      = idx & 7;
  int lane   = (idx >> 3) & 31;
  int kt     = (idx >> 8) % Ktiles;
  int coTile = idx / (256 * Ktiles);
  int co     = coTile * 16 + (lane & 15);
  int kbase  = (j < 4 ? 0 : 16) + ((lane >> 4) << 3);
  int kloc   = kbase + ((j & 3) << 1);
  unsigned int d = 0;
#pragma unroll
  for (int s = 0; s < 2; ++s) {
    int k  = kt * 32 + kloc + s;
    int ci = k % CinPad, r = k / CinPad;
    unsigned short hv = 0;
    if (co < Cout && ci < CinReal)
      hv = f32_to_bf16(W[((size_t)co * CinReal + ci) * ksq + r]);
    d |= (unsigned int)hv << (16 * s);
  }
  dst[idx] = d;
}

// ---------------------------------------------------------------------------
// 3x3 SAME conv, implicit GEMM, bf16 in / f32 acc (v_wmma_f32_16x16x32_bf16).
// Block = 256 threads = 8 waves; block owns one (n,h) 128-px half-row and ALL
// NT cout tiles (NT accumulators).  Per (kh, ci-block): stage 32x(128+halo)
// row tile into LDS with global_load_async_to_lds_b128 (double-buffered,
// s_wait_asynccnt pipelining), then 3 kw taps x NT WMMAs from one tile.
// Gather is 16 ds_load_u16 at immediate offsets (no per-element div/mod).
// ---------------------------------------------------------------------------
template <int NT>
__global__ void conv3x3_wmma(const unsigned short* __restrict__ X, int CB, int inCtot,
                             const unsigned int* __restrict__ Aw,
                             const unsigned short* __restrict__ resid, int residCtot,
                             unsigned short* __restrict__ out, int outCtot,
                             unsigned short* __restrict__ out2, int out2Ctot,
                             float* __restrict__ outF32, int coMax, int doRelu) {
  // row stride 136 ushorts = 272B (16B-aligned rows); slots: 0..127 body,
  // 128 = right halo (gw=wb+128), 129 = left halo (gw=wb-1)
  __shared__ __align__(16) unsigned short tile[2][32][136];
  const int tid  = threadIdx.x;
  const int wave = tid >> 5, lane = tid & 31;
  const int row  = blockIdx.x >> 1;           // 0..2047
  const int n    = row >> 8, h = row & 255;
  const int wb   = (blockIdx.x & 1) << 7;     // 0 / 128
  const int halfsel = lane >> 4;
  const int pxl     = (wave << 4) + (lane & 15);
  const int Ktiles  = 9 * CB;

  floatx8 acc[NT];
#pragma unroll
  for (int ct = 0; ct < NT; ++ct) acc[ct] = (floatx8){};

  auto stage = [&](int bb, int gh, int cb) {
    const unsigned short* src =
        X + ((size_t)(n * inCtot + cb * 32)) * SPA + gh * 256;
#pragma unroll
    for (int q = 0; q < 2; ++q) {
      int l = tid + q * 256;                  // 512 b128 loads total
      int rw = l >> 4, seg = l & 15;
      unsigned int loff = lds_off32(&tile[bb][rw][seg * 8]);
      unsigned long long ga =
          (unsigned long long)(uintptr_t)(src + (size_t)rw * SPA + wb + seg * 8);
      asm volatile("global_load_async_to_lds_b128 %0, %1, off"
                   :: "v"(loff), "v"(ga) : "memory");
    }
    if (tid < 64) {                           // 2 halo columns
      int rw   = tid & 31;
      int gw   = (tid < 32) ? (wb - 1) : (wb + 128);
      int slot = (tid < 32) ? 129 : 128;
      unsigned short v = 0;
      if (gw >= 0 && gw < 256) v = src[(size_t)rw * SPA + gw];
      tile[bb][rw][slot] = v;
    }
  };

  auto compute = [&](int bb, int kh, int cb) {
#pragma unroll
    for (int kw = 0; kw < 3; ++kw) {
      int xidx = pxl + kw - 1;
      if (xidx < 0) xidx = 129;               // left halo slot
      const unsigned short* bp = &tile[bb][halfsel * 8][xidx];
      U16x16 b;
#pragma unroll
      for (int j = 0; j < 16; ++j) b.u[j] = bp[(j + (j >= 8 ? 8 : 0)) * 136];
      const int kt = (kh * 3 + kw) * CB + cb;
#pragma unroll
      for (int ct = 0; ct < NT; ++ct) {
        U16x16 a;
        const u32x4* ap =
            (const u32x4*)(Aw + (((size_t)ct * Ktiles + kt) * 32 + lane) * 8);
        *(u32x4*)&a.u[0] = ap[0];
        *(u32x4*)&a.u[8] = ap[1];
        acc[ct] = __builtin_amdgcn_wmma_f32_16x16x32_bf16(
            false, a.v, false, b.v, (short)0, acc[ct], false, false);
      }
    }
  };

  // software pipeline: stage chunk i+1 while computing chunk i
  int pkh = -1, pcb = 0, pbb = 0, buf = 0;
  for (int kh = 0; kh < 3; ++kh) {
    int gh = h + kh - 1;
    if (gh < 0 || gh > 255) continue;         // uniform: zero rows contribute 0
    for (int cb = 0; cb < CB; ++cb) {
      stage(buf, gh, cb);
      asm volatile("s_wait_asynccnt 0x2" ::: "memory");  // prev batch landed
      __syncthreads();
      if (pkh >= 0) compute(pbb, pkh, pcb);
      __syncthreads();
      pkh = kh; pcb = cb; pbb = buf; buf ^= 1;
    }
  }
  asm volatile("s_wait_asynccnt 0x0" ::: "memory");
  __syncthreads();
  compute(pbb, pkh, pcb);

  // epilogue: fused residual-add / relu, bf16 (+optional 2nd dest) or f32
  const int wout = wb + pxl;
#pragma unroll
  for (int ct = 0; ct < NT; ++ct) {
#pragma unroll
    for (int r = 0; r < 8; ++r) {
      int co = ct * 16 + r + halfsel * 8;
      float val = acc[ct][r];
      if (resid)
        val += bf16_to_f32(resid[((size_t)(n * residCtot + co)) * SPA + h * 256 + wout]);
      if (doRelu) val = fmaxf(val, 0.f);
      if (outF32) {
        if (co < coMax)
          outF32[((size_t)(n * coMax + co)) * SPA + h * 256 + wout] = val;
      } else {
        unsigned short ov = f32_to_bf16(val);
        out[((size_t)(n * outCtot + co)) * SPA + h * 256 + wout] = ov;
        if (out2)
          out2[((size_t)(n * out2Ctot + co)) * SPA + h * 256 + wout] = ov;
      }
    }
  }
}

// ---------------------------------------------------------------------------
// 1x1 conv (pure channel GEMM), same async double-buffered pipeline.
// ---------------------------------------------------------------------------
template <int NT>
__global__ void conv1x1_wmma(const unsigned short* __restrict__ X, int CB, int inCtot,
                             const unsigned int* __restrict__ Aw,
                             const unsigned short* __restrict__ resid, int residCtot,
                             unsigned short* __restrict__ out, int outCtot,
                             unsigned short* __restrict__ out2, int out2Ctot,
                             int doRelu) {
  __shared__ __align__(16) unsigned short tile[2][32][128];
  const int tid  = threadIdx.x;
  const int wave = tid >> 5, lane = tid & 31;
  const int row  = blockIdx.x >> 1;
  const int n    = row >> 8, h = row & 255;
  const int wb   = (blockIdx.x & 1) << 7;
  const int halfsel = lane >> 4;
  const int pxl     = (wave << 4) + (lane & 15);

  floatx8 acc[NT];
#pragma unroll
  for (int ct = 0; ct < NT; ++ct) acc[ct] = (floatx8){};

  auto stage = [&](int bb, int cb) {
    const unsigned short* src =
        X + ((size_t)(n * inCtot + cb * 32)) * SPA + h * 256 + wb;
#pragma unroll
    for (int q = 0; q < 2; ++q) {
      int l = tid + q * 256;
      int rw = l >> 4, seg = l & 15;
      unsigned int loff = lds_off32(&tile[bb][rw][seg * 8]);
      unsigned long long ga =
          (unsigned long long)(uintptr_t)(src + (size_t)rw * SPA + seg * 8);
      asm volatile("global_load_async_to_lds_b128 %0, %1, off"
                   :: "v"(loff), "v"(ga) : "memory");
    }
  };

  auto compute = [&](int bb, int cb) {
    U16x16 b;
    const unsigned short* bp = &tile[bb][halfsel * 8][pxl];
#pragma unroll
    for (int j = 0; j < 16; ++j) b.u[j] = bp[(j + (j >= 8 ? 8 : 0)) * 128];
#pragma unroll
    for (int ct = 0; ct < NT; ++ct) {
      U16x16 a;
      const u32x4* ap =
          (const u32x4*)(Aw + (((size_t)ct * CB + cb) * 32 + lane) * 8);
      *(u32x4*)&a.u[0] = ap[0];
      *(u32x4*)&a.u[8] = ap[1];
      acc[ct] = __builtin_amdgcn_wmma_f32_16x16x32_bf16(
          false, a.v, false, b.v, (short)0, acc[ct], false, false);
    }
  };

  int pcb = -1, pbb = 0, buf = 0;
  for (int cb = 0; cb < CB; ++cb) {
    stage(buf, cb);
    asm volatile("s_wait_asynccnt 0x2" ::: "memory");
    __syncthreads();
    if (pcb >= 0) compute(pbb, pcb);
    __syncthreads();
    pcb = cb; pbb = buf; buf ^= 1;
  }
  asm volatile("s_wait_asynccnt 0x0" ::: "memory");
  __syncthreads();
  compute(pbb, pcb);

  const int wout = wb + pxl;
#pragma unroll
  for (int ct = 0; ct < NT; ++ct) {
#pragma unroll
    for (int r = 0; r < 8; ++r) {
      int co = ct * 16 + r + halfsel * 8;
      float val = acc[ct][r];
      if (resid)
        val += bf16_to_f32(resid[((size_t)(n * residCtot + co)) * SPA + h * 256 + wout]);
      if (doRelu) val = fmaxf(val, 0.f);
      unsigned short ov = f32_to_bf16(val);
      out[((size_t)(n * outCtot + co)) * SPA + h * 256 + wout] = ov;
      if (out2)
        out2[((size_t)(n * out2Ctot + co)) * SPA + h * 256 + wout] = ov;
    }
  }
}

// ---------------------------------------------------------------------------
// ADMM element-wise / reduction kernels
// ---------------------------------------------------------------------------
__global__ void fill_zero_u32(unsigned int* __restrict__ p, int total) {
  int i = blockIdx.x * 256 + threadIdx.x;
  if (i < total) p[i] = 0;
}

__global__ void mean24(const float* __restrict__ J, float* __restrict__ means) {
  __shared__ float sm[256];
  const float* p = J + (size_t)blockIdx.x * SPA;
  float s = 0.f;
  for (int i = threadIdx.x; i < SPA; i += 256) s += p[i];
  sm[threadIdx.x] = s; __syncthreads();
  for (int o = 128; o > 0; o >>= 1) {
    if (threadIdx.x < o) sm[threadIdx.x] += sm[threadIdx.x + o];
    __syncthreads();
  }
  if (threadIdx.x == 0) means[blockIdx.x] = sm[0] * (1.0f / SPA);
}

// per-(n,pix): stable channel sort by mean, J_m/J_s adjustment -> Jw
__global__ void chanstat(const float* __restrict__ J, const float* __restrict__ means,
                         float* __restrict__ Jw) {
  int i = blockIdx.x * 256 + threadIdx.x;
  if (i >= NB * SPA) return;
  int n = i / SPA, pix = i % SPA;
  float mv[3] = {means[n * 3 + 0], means[n * 3 + 1], means[n * 3 + 2]};
  int o[3] = {0, 1, 2};
  if (mv[o[1]] < mv[o[0]]) { int t = o[0]; o[0] = o[1]; o[1] = t; }
  if (mv[o[2]] < mv[o[1]]) { int t = o[1]; o[1] = o[2]; o[2] = t; }
  if (mv[o[1]] < mv[o[0]]) { int t = o[0]; o[0] = o[1]; o[1] = t; }
  float jv[3];
  jv[0] = J[((size_t)n * 3 + 0) * SPA + pix];
  jv[1] = J[((size_t)n * 3 + 1) * SPA + pix];
  jv[2] = J[((size_t)n * 3 + 2) * SPA + pix];
  float Jl = jv[o[2]], lbar = mv[o[2]];
  float ov[3];
  ov[o[2]] = Jl;
  ov[o[1]] = jv[o[1]] + (lbar - mv[o[1]]) * Jl;
  ov[o[0]] = jv[o[0]] + (lbar - mv[o[0]]) * Jl;
  Jw[((size_t)n * 3 + 0) * SPA + pix] = ov[0];
  Jw[((size_t)n * 3 + 1) * SPA + pix] = ov[1];
  Jw[((size_t)n * 3 + 2) * SPA + pix] = ov[2];
}

// B pre-image fused into its spatial-mean reduction
__global__ void b_reduce(const float* __restrict__ Bp, const float* __restrict__ Jw,
                         const float* __restrict__ t, const float* __restrict__ I,
                         float* __restrict__ Bmean) {
  __shared__ float sm[256];
  size_t base = (size_t)blockIdx.x * SPA;
  float s = 0.f;
  for (int i = threadIdx.x; i < SPA; i += 256) {
    float tv = t[base + i], iv = I[base + i];
    float omt = 1.0f - tv;
    float v = (ETA2f * Bp[base + i] - ETA0f * (Jw[base + i] * tv - iv) * omt)
            / (ETA0f * omt * omt + ETA2f);
    s += v;
  }
  sm[threadIdx.x] = s; __syncthreads();
  for (int o = 128; o > 0; o >>= 1) {
    if (threadIdx.x < o) sm[threadIdx.x] += sm[threadIdx.x + o];
    __syncthreads();
  }
  if (threadIdx.x == 0) Bmean[blockIdx.x] = sm[0] * (1.0f / SPA);
}

__global__ void bcast_B(const float* __restrict__ Bmean, float* __restrict__ oB, int total) {
  int i = blockIdx.x * 256 + threadIdx.x;
  if (i >= total) return;
  oB[i] = Bmean[i / SPA];
}

// t update + 1x1 t1d mix + channel tile + bf16 RDN input (t + LagT/rho4)
// Xin is 32-channel zero-padded (channels 3..31 pre-zeroed) for the r-major
// WMMA path of sfe1.
__global__ void t_update(const float* __restrict__ t_p, const float* __restrict__ AuxT,
                         const float* __restrict__ LagT, const float* __restrict__ I,
                         const float* __restrict__ Jw, const float* __restrict__ Bmean,
                         const float* __restrict__ t1w, const float* __restrict__ rhos,
                         float* __restrict__ oT, unsigned short* __restrict__ Xin) {
  int i = blockIdx.x * 256 + threadIdx.x;
  if (i >= NB * SPA) return;
  int n = i / SPA, pix = i % SPA;
  float rho4 = rhos[3];
  float t1 = 0.f;
#pragma unroll
  for (int c = 0; c < 3; ++c) {
    size_t id = ((size_t)n * 3 + c) * SPA + pix;
    float B = Bmean[n * 3 + c];
    float jb = Jw[id] - B;
    float tc = (ETA2f * t_p[id] + rho4 * AuxT[id] - LagT[id] - ETA0f * (B - I[id]) * jb)
             / (ETA0f * jb * jb + ETA1f + rho4);
    t1 += t1w[c] * tc;
  }
#pragma unroll
  for (int c = 0; c < 3; ++c) {
    size_t id = ((size_t)n * 3 + c) * SPA + pix;
    oT[id] = t1;
    Xin[((size_t)n * 32 + c) * SPA + pix] = f32_to_bf16(t1 + LagT[id] / rho4);
  }
}

// J, DCP, BCP, Lag_J in one pass
__global__ void j_update(const float* __restrict__ oT, const float* __restrict__ I,
                         const float* __restrict__ AuxJ, const float* __restrict__ LagJ,
                         const float* __restrict__ DCPi, const float* __restrict__ BCPi,
                         const float* __restrict__ Bmean, const float* __restrict__ rhos,
                         float* __restrict__ oJ, float* __restrict__ oD,
                         float* __restrict__ oBC, float* __restrict__ oLJ, int total) {
  int i = blockIdx.x * 256 + threadIdx.x;
  if (i >= total) return;
  int nc = i / SPA;
  float rho1 = rhos[0], rho2 = rhos[1], rho3 = rhos[2], rho4 = rhos[3],
        rho5 = rhos[4], rho6 = rhos[5];
  float tv = oT[i], B = Bmean[nc];
  float num = tv * (I[i] - B * (1.0f - tv)) + rho3 * AuxJ[i] - LagJ[i]
            + rho5 * DCPi[i] - rho6 * BCPi[i] + rho6;
  float Jn = ETA0f * num / (ETA0f * tv * tv + rho3 + rho5 + rho6);
  oJ[i]  = Jn;
  oD[i]  = (rho1 * DCPi[i] + rho4 * Jn) / (rho1 + rho4);
  oBC[i] = (rho2 * BCPi[i] - rho5 * Jn + rho5) / (rho2 + rho5);
  oLJ[i] = LagJ[i] + rho3 * (Jn - AuxJ[i]);
}

__global__ void lagt_update(const float* __restrict__ LagT, const float* __restrict__ oT,
                            const float* __restrict__ oAT, const float* __restrict__ rhos,
                            float* __restrict__ oLT, int total) {
  int i = blockIdx.x * 256 + threadIdx.x;
  if (i >= total) return;
  oLT[i] = LagT[i] + rhos[3] * (oT[i] - oAT[i]);
}

// channel-min -> (N,1,H,W)
__global__ void minc3(const float* __restrict__ x, float* __restrict__ out) {
  int i = blockIdx.x * 256 + threadIdx.x;
  if (i >= NB * SPA) return;
  int n = i / SPA, pix = i % SPA;
  size_t b = (size_t)n * 3 * SPA + pix;
  out[i] = fminf(x[b], fminf(x[b + SPA], x[b + 2 * (size_t)SPA]));
}

// separable 35x35 min (SAME, +inf pad => clamp range). patch_size=35 is static
// in the JAX trace (lax.reduce_window window shape), so it is compile-time here.
__global__ void minpool_h(const float* __restrict__ in, float* __restrict__ out) {
  int i = blockIdx.x * 256 + threadIdx.x;
  if (i >= NB * SPA) return;
  int n = i / SPA, rem = i % SPA, h = rem >> 8, w = rem & 255;
  const float* row = in + (size_t)n * SPA + h * 256;
  int lo = imax(w - 17, 0), hi = imin(w + 17, 255);
  float m = __builtin_inff();
  for (int x = lo; x <= hi; ++x) m = fminf(m, row[x]);
  out[i] = m;
}
__global__ void minpool_v_soft(const float* __restrict__ in, float* __restrict__ out,
                               const float* __restrict__ rhos, int rhoIdx, float eta) {
  int i = blockIdx.x * 256 + threadIdx.x;
  if (i >= NB * SPA) return;
  int n = i / SPA, rem = i % SPA, h = rem >> 8, w = rem & 255;
  int lo = imax(h - 17, 0), hi = imin(h + 17, 255);
  float m = __builtin_inff();
  const float* col = in + (size_t)n * SPA + w;
  for (int y = lo; y <= hi; ++y) m = fminf(m, col[y * 256]);
  float thr = eta / rhos[rhoIdx];
  float s = fmaxf(fabsf(m) - thr, 0.f);
  out[i] = (m > 0.f) ? s : ((m < 0.f) ? -s : 0.f);
}

__global__ void write_rho3(const float* __restrict__ rhos, float* __restrict__ o) {
  if (threadIdx.x == 0 && blockIdx.x == 0) o[0] = rhos[2];
}

// ---------------------------------------------------------------------------
// Host orchestration
// ---------------------------------------------------------------------------
extern "C" void kernel_launch(void* const* d_in, const int* in_sizes, int n_in,
                              void* d_out, int out_size, void* d_ws, size_t ws_size,
                              hipStream_t stream) {
  (void)in_sizes; (void)n_in; (void)out_size; (void)ws_size;

  const float* I    = (const float*)d_in[0];
  const float* t_p  = (const float*)d_in[1];
  const float* B_p  = (const float*)d_in[2];
  const float* t_in = (const float*)d_in[4];
  const float* J_in = (const float*)d_in[5];
  const float* AuxJ = (const float*)d_in[6];
  const float* AuxT = (const float*)d_in[7];
  const float* LagJ = (const float*)d_in[8];
  const float* LagT = (const float*)d_in[9];
  const float* DCPi = (const float*)d_in[10];
  const float* BCPi = (const float*)d_in[11];
  const float* rhos = (const float*)d_in[15];
  const float* t1w  = (const float*)d_in[16];
  // rdn_params flattened in make_rdn_params insertion order -> d_in[17..29]

  // ---- workspace layout (bytes) ----
  char* ws = (char*)d_ws;
  float*        means = (float*)(ws + 0);
  float*        Bmean = (float*)(ws + 256);
  unsigned int* Wpk   = (unsigned int*)(ws + 512);
  size_t off = 512 + (1u << 20);
  float* Jw   = (float*)(ws + off); off += (size_t)NB * 3 * SPA * 4;
  float* minU = (float*)(ws + off); off += (size_t)NB * SPA * 4;
  float* minV = (float*)(ws + off); off += (size_t)NB * SPA * 4;
  float* ptmp = (float*)(ws + off); off += (size_t)NB * SPA * 4;
  unsigned short* Xin = (unsigned short*)(ws + off); off += (size_t)NB * 32  * SPA * 2;
  unsigned short* F1  = (unsigned short*)(ws + off); off += (size_t)NB * 64  * SPA * 2;
  unsigned short* Hb  = (unsigned short*)(ws + off); off += (size_t)NB * 160 * SPA * 2;
  unsigned short* FE  = (unsigned short*)(ws + off); off += (size_t)NB * 128 * SPA * 2;
  unsigned short* G1  = (unsigned short*)(ws + off); off += (size_t)NB * 64  * SPA * 2;
  unsigned short* G2  = (unsigned short*)(ws + off); off += (size_t)NB * 64  * SPA * 2;

  // ---- output layout ----
  float* out = (float*)d_out;
  const size_t E3 = (size_t)NB * 3 * SPA;    // 1,572,864
  const int    E1 = NB * SPA;                // 524,288
  float* oB  = out;
  float* oT  = out + E3;
  float* oJ  = out + 2 * E3;
  float* oAJ = out + 3 * E3;
  float* oAT = out + 4 * E3;
  float* oLJ = out + 5 * E3;
  float* oLT = out + 6 * E3;
  float* oMU = out + 7 * E3;
  float* oMV = oMU + E1;
  float* oD  = oMV + E1;
  float* oBC = oD + E3;
  float* oR3 = oBC + E3;

  // ---- RDN layer table + weight repack (r-major K) ----
  struct LDesc { int Cout, CinReal, CinPad, ks; };
  const LDesc L[13] = {
    {64, 3, 32, 3}, {64, 64, 64, 3},                                   // sfe1, sfe2
    {32, 64, 64, 3}, {32, 96, 96, 3}, {32, 128, 128, 3}, {64, 160, 160, 1}, // rdb0
    {32, 64, 64, 3}, {32, 96, 96, 3}, {32, 128, 128, 3}, {64, 160, 160, 1}, // rdb1
    {64, 128, 128, 1}, {64, 64, 64, 3},                                // gff1, gff2
    {3, 64, 64, 3}                                                     // out
  };
  size_t wofs[13]; size_t cur = 0;
  for (int i = 0; i < 13; ++i) {
    wofs[i] = cur;
    int ksq = L[i].ks * L[i].ks;
    cur += (size_t)((L[i].Cout + 15) / 16) * ((L[i].CinPad * ksq) / 32) * 256;
  }
  for (int i = 0; i < 13; ++i) {
    int ksq = L[i].ks * L[i].ks;
    int Kt = (L[i].CinPad * ksq) / 32, Ct = (L[i].Cout + 15) / 16;
    int tot = Ct * Kt * 256;
    repack_w<<<(tot + 255) / 256, 256, 0, stream>>>(
        (const float*)d_in[17 + i], L[i].Cout, L[i].CinReal, L[i].CinPad,
        ksq, Wpk + wofs[i], tot);
  }

  const int GE1 = (E1 + 255) / 256;
  const int GE3 = (int)((E3 + 255) / 256);

  // ---- ADMM scalar part ----
  {
    int tot = NB * 32 * SPA / 2;   // zero 32-ch padded RDN input (dwords)
    fill_zero_u32<<<(tot + 255) / 256, 256, 0, stream>>>((unsigned int*)Xin, tot);
  }
  mean24<<<24, 256, 0, stream>>>(J_in, means);
  chanstat<<<GE1, 256, 0, stream>>>(J_in, means, Jw);
  b_reduce<<<24, 256, 0, stream>>>(B_p, Jw, t_in, I, Bmean);
  bcast_B<<<GE3, 256, 0, stream>>>(Bmean, oB, (int)E3);
  t_update<<<GE1, 256, 0, stream>>>(t_p, AuxT, LagT, I, Jw, Bmean, t1w, rhos, oT, Xin);
  j_update<<<GE3, 256, 0, stream>>>(oT, I, AuxJ, LagJ, DCPi, BCPi, Bmean, rhos,
                                    oJ, oD, oBC, oLJ, (int)E3);

  // ---- RDN forward (bf16 WMMA, async-LDS pipeline) ----
  auto c3 = [&](int li, const unsigned short* X, int inCtot,
                const unsigned short* resid, int residCtot,
                unsigned short* o1, int o1Ct, unsigned short* o2, int o2Ct,
                float* oF, int coMax, int relu) {
    int CB = L[li].CinPad / 32;
    int Ct = (L[li].Cout + 15) / 16;
    const unsigned int* A = Wpk + wofs[li];
    switch (Ct) {
      case 1: conv3x3_wmma<1><<<4096, 256, 0, stream>>>(X, CB, inCtot, A, resid, residCtot, o1, o1Ct, o2, o2Ct, oF, coMax, relu); break;
      case 2: conv3x3_wmma<2><<<4096, 256, 0, stream>>>(X, CB, inCtot, A, resid, residCtot, o1, o1Ct, o2, o2Ct, oF, coMax, relu); break;
      default: conv3x3_wmma<4><<<4096, 256, 0, stream>>>(X, CB, inCtot, A, resid, residCtot, o1, o1Ct, o2, o2Ct, oF, coMax, relu); break;
    }
  };
  auto c1 = [&](int li, const unsigned short* X, int inCtot,
                const unsigned short* resid, int residCtot,
                unsigned short* o1, int o1Ct, unsigned short* o2, int o2Ct, int relu) {
    int CB = L[li].CinPad / 32;
    int Ct = (L[li].Cout + 15) / 16;
    const unsigned int* A = Wpk + wofs[li];
    switch (Ct) {
      case 1: conv1x1_wmma<1><<<4096, 256, 0, stream>>>(X, CB, inCtot, A, resid, residCtot, o1, o1Ct, o2, o2Ct, relu); break;
      case 2: conv1x1_wmma<2><<<4096, 256, 0, stream>>>(X, CB, inCtot, A, resid, residCtot, o1, o1Ct, o2, o2Ct, relu); break;
      default: conv1x1_wmma<4><<<4096, 256, 0, stream>>>(X, CB, inCtot, A, resid, residCtot, o1, o1Ct, o2, o2Ct, relu); break;
    }
  };

  c3(0, Xin, 32, nullptr, 0, F1, 64, nullptr, 0, nullptr, 0, 0);        // sfe1
  c3(1, F1, 64,  nullptr, 0, Hb, 160, nullptr, 0, nullptr, 0, 0);       // sfe2 -> xd = H[0:64]
  // RDB 0 (concat via channel-offset stores into Hb)
  c3(2, Hb, 160, nullptr, 0, Hb + (size_t)64  * SPA, 160, nullptr, 0, nullptr, 0, 1);
  c3(3, Hb, 160, nullptr, 0, Hb + (size_t)96  * SPA, 160, nullptr, 0, nullptr, 0, 1);
  c3(4, Hb, 160, nullptr, 0, Hb + (size_t)128 * SPA, 160, nullptr, 0, nullptr, 0, 1);
  c1(5, Hb, 160, Hb, 160, Hb, 160, FE, 128, 0);                          // lff0 + xin -> xd, feats[0]
  // RDB 1
  c3(6, Hb, 160, nullptr, 0, Hb + (size_t)64  * SPA, 160, nullptr, 0, nullptr, 0, 1);
  c3(7, Hb, 160, nullptr, 0, Hb + (size_t)96  * SPA, 160, nullptr, 0, nullptr, 0, 1);
  c3(8, Hb, 160, nullptr, 0, Hb + (size_t)128 * SPA, 160, nullptr, 0, nullptr, 0, 1);
  c1(9, Hb, 160, Hb, 160, Hb, 160, FE + (size_t)64 * SPA, 128, 0);       // lff1 -> feats[1]
  c1(10, FE, 128, nullptr, 0, G1, 64, nullptr, 0, 0);                    // gff1
  c3(11, G1, 64, F1, 64, G2, 64, nullptr, 0, nullptr, 0, 0);             // gff2 + f1
  c3(12, G2, 64, nullptr, 0, nullptr, 0, nullptr, 0, oAT, 3, 0);         // out -> Aux_t (f32)

  // ---- tail ----
  lagt_update<<<GE3, 256, 0, stream>>>(LagT, oT, oAT, rhos, oLT, (int)E3);
  hipMemcpyAsync(oAJ, AuxJ, E3 * sizeof(float), hipMemcpyDeviceToDevice, stream);

  minc3<<<GE1, 256, 0, stream>>>(oD, minU);
  minc3<<<GE1, 256, 0, stream>>>(oBC, minV);
  minpool_h<<<GE1, 256, 0, stream>>>(minU, ptmp);
  minpool_v_soft<<<GE1, 256, 0, stream>>>(ptmp, oMU, rhos, 0, ETA3f);
  minpool_h<<<GE1, 256, 0, stream>>>(minV, ptmp);
  minpool_v_soft<<<GE1, 256, 0, stream>>>(ptmp, oMV, rhos, 1, ETA4f);
  write_rho3<<<1, 32, 0, stream>>>(rhos, oR3);
}
